// HDGT_encoder_5119601016849
// MI455X (gfx1250) — matrix operand order repaired and checked by reading, not compiled, not run on hardware
//
#include <hip/hip_runtime.h>
#include <cmath>

typedef __attribute__((ext_vector_type(16))) __bf16 v16bf;
typedef __attribute__((ext_vector_type(8)))  __bf16 bf16x8;
typedef __attribute__((ext_vector_type(8)))  float  v8f;

#define NA 4096   // agents
#define DM 256    // model dim

// ---------------------------------------------------------------------------
// Pack a row-major f32 weight [K, Nc] into WMMA B-fragment order (bf16):
// out[((nt*KT + kt)*32 + lane)*16 + i] = W[kt*32 + (lane>>4)*8 + i + (i&8)][nt*16 + (lane&15)]
// so a wave reads its 32x16 B tile as one contiguous 32B-per-lane load.
// ---------------------------------------------------------------------------
__global__ void pack_b_kernel(const float* __restrict__ W, __bf16* __restrict__ out,
                              int KT, int Nc) {
  int tile = blockIdx.x;          // tile = nt*KT + kt
  int kt = tile % KT;
  int nt = tile / KT;
  int lane = threadIdx.x;
  const float* src = W + (size_t)(kt * 32 + (lane >> 4) * 8) * Nc + nt * 16 + (lane & 15);
  __bf16* dst = out + ((size_t)tile * 32 + lane) * 16;
#pragma unroll
  for (int i = 0; i < 16; ++i) dst[i] = (__bf16)src[(size_t)(i + (i & 8)) * Nc];
}

// ---------------------------------------------------------------------------
// Generic GEMM: C[M,Nc] f32 = A[M,K] bf16 (row-major, lda) x Bpacked.
// 128 threads = 4 waves; each wave owns one 16-col N-tile and FOUR 16-row
// M-tiles (64 rows per block), so each B fragment register set feeds 4 WMMAs
// (4x less weight traffic through WGP$/L2). grid = (M/64, Nc/64).
// Epilogue: optional bias, optional ReLU, optional per-row type-match store
// (implements the reference's one-hot select without gather/scatter).
// ---------------------------------------------------------------------------
__global__ __launch_bounds__(128)
void gemm_bf16_kernel(const __bf16* __restrict__ A, int lda,
                      const __bf16* __restrict__ Bp, int KT,
                      const float* __restrict__ bias,
                      float* __restrict__ C, int ldc,
                      int M, int act,
                      const int* __restrict__ typ, int tmatch) {
  const int wave = threadIdx.x >> 5;
  const int lane = threadIdx.x & 31;
  const int m0 = blockIdx.x * 64;
  const int nt = blockIdx.y * 4 + wave;
  const int half = lane >> 4;

  v8f acc[4];
#pragma unroll
  for (int mt = 0; mt < 4; ++mt) acc[mt] = (v8f){0.f,0.f,0.f,0.f,0.f,0.f,0.f,0.f};

  const __bf16* bptr = Bp + ((size_t)nt * KT) * 512 + (size_t)lane * 16;
  const __bf16* abase = A + (size_t)half * 8;
  for (int kt = 0; kt < KT; ++kt) {
    v16bf bv = *(const v16bf*)(bptr);
#pragma unroll
    for (int mt = 0; mt < 4; ++mt) {
      int arow = m0 + mt * 16 + (lane & 15);
      if (arow >= M) arow = M - 1;
      const __bf16* ap = abase + (size_t)arow * lda + kt * 32;
      union { v16bf v; bf16x8 h[2]; } au;
      au.h[0] = *(const bf16x8*)(ap);
      au.h[1] = *(const bf16x8*)(ap + 16);
      acc[mt] = __builtin_amdgcn_wmma_f32_16x16x32_bf16(false, au.v, false, bv,
                                                        (short)0, acc[mt], false, false);
    }
    bptr += 512;
  }
  const int col = nt * 16 + (lane & 15);
  const float badd = bias ? bias[col] : 0.f;
#pragma unroll
  for (int mt = 0; mt < 4; ++mt) {
    const int rbase = m0 + mt * 16 + half * 8;
#pragma unroll
    for (int r = 0; r < 8; ++r) {
      int row = rbase + r;
      if (row >= M) continue;
      if (typ && typ[row] != tmatch) continue;
      float v = acc[mt][r] + badd;
      if (act == 1) v = fmaxf(v, 0.f);
      C[(size_t)row * ldc + col] = v;
    }
  }
}

// ---------------------------------------------------------------------------
// Fused per-agent KV projection + attention. One block (256 thr / 8 waves)
// per agent: single-pass register LN (each neighbor row read from HBM exactly
// once) -> bf16 LDS, WMMA [Lp,256]x[256,512] KV GEMM into LDS with kt-outer
// loop so each B fragment is reused across both M-tiles, masked softmax per
// head (wave shuffles), PV -> o[n, 256].
// Agent type is uniform per block, so typed KV weights cost no extra FLOPs.
// ---------------------------------------------------------------------------
__global__ __launch_bounds__(256)
void attn_kv_kernel(const float* __restrict__ neigh, const int* __restrict__ mask,
                    const float* __restrict__ lng, const float* __restrict__ lnb,
                    int typed_ln,
                    const __bf16* __restrict__ Bp, size_t bstride,
                    const float* __restrict__ q_sel, int qoff,
                    const int* __restrict__ agent_type,
                    float* __restrict__ o_out, int L, float scale) {
  __shared__ __attribute__((aligned(16))) __bf16 lds_a[32 * 256];   // 16 KB
  __shared__ __attribute__((aligned(16))) __bf16 lds_kv[32 * 512];  // 32 KB
  __shared__ float lds_p[8 * 32];

  const int n = blockIdx.x;
  const int tid = threadIdx.x;
  const int wave = tid >> 5;
  const int lane = tid & 31;
  const int t = agent_type[n];
  const int Lp = (L + 15) & ~15;

  // ---- single-pass LN: load row into registers, butterfly stats, store bf16
  const float* g = lng + (typed_ln ? t * 256 : 0);
  const float* b = lnb + (typed_ln ? t * 256 : 0);
  float gv[8], bv8[8];
#pragma unroll
  for (int j = 0; j < 8; ++j) { gv[j] = g[lane * 8 + j]; bv8[j] = b[lane * 8 + j]; }

  for (int r = wave; r < Lp; r += 8) {
    if (r < L) {
      const float* xr = neigh + ((size_t)n * L + r) * 256;
      float v[8];
      float s = 0.f, ss = 0.f;
#pragma unroll
      for (int j = 0; j < 8; ++j) { v[j] = xr[lane * 8 + j]; s += v[j]; ss += v[j] * v[j]; }
#pragma unroll
      for (int o = 16; o > 0; o >>= 1) { s += __shfl_xor(s, o, 32); ss += __shfl_xor(ss, o, 32); }
      const float m = s * (1.f / 256.f);
      const float rs = rsqrtf(ss * (1.f / 256.f) - m * m + 1e-5f);
#pragma unroll
      for (int j = 0; j < 8; ++j)
        lds_a[r * 256 + lane * 8 + j] = (__bf16)((v[j] - m) * rs * gv[j] + bv8[j]);
    } else {
#pragma unroll
      for (int j = 0; j < 8; ++j) lds_a[r * 256 + lane * 8 + j] = (__bf16)0.f;
    }
  }
  __syncthreads();

  // ---- WMMA KV GEMM: [Lp x 256] x [256 x 512] -> lds_kv (bf16), KT=8, NT=32
  const __bf16* Bpt = Bp + (size_t)t * bstride;
  const int MT = Lp >> 4;            // 1 or 2, uniform across the block
  const int half = lane >> 4;
  for (int nti = 0; nti < 4; ++nti) {
    const int nt = wave + nti * 8;
    v8f acc0 = {0.f,0.f,0.f,0.f,0.f,0.f,0.f,0.f};
    v8f acc1 = {0.f,0.f,0.f,0.f,0.f,0.f,0.f,0.f};
#pragma unroll
    for (int kt = 0; kt < 8; ++kt) {
      v16bf bv = *(const v16bf*)(Bpt + (((size_t)nt * 8 + kt) * 32 + lane) * 16);
      {
        union { v16bf v; bf16x8 h[2]; } au;
        const __bf16* ap = &lds_a[(lane & 15) * 256 + kt * 32 + half * 8];
        au.h[0] = *(const bf16x8*)ap;
        au.h[1] = *(const bf16x8*)(ap + 16);
        acc0 = __builtin_amdgcn_wmma_f32_16x16x32_bf16(false, au.v, false, bv,
                                                       (short)0, acc0, false, false);
      }
      if (MT == 2) {
        union { v16bf v; bf16x8 h[2]; } au;
        const __bf16* ap = &lds_a[(16 + (lane & 15)) * 256 + kt * 32 + half * 8];
        au.h[0] = *(const bf16x8*)ap;
        au.h[1] = *(const bf16x8*)(ap + 16);
        acc1 = __builtin_amdgcn_wmma_f32_16x16x32_bf16(false, au.v, false, bv,
                                                       (short)0, acc1, false, false);
      }
    }
    const int col = nt * 16 + (lane & 15);
    const int rbase = half * 8;
#pragma unroll
    for (int r = 0; r < 8; ++r) lds_kv[(rbase + r) * 512 + col] = (__bf16)acc0[r];
    if (MT == 2) {
#pragma unroll
      for (int r = 0; r < 8; ++r) lds_kv[(16 + rbase + r) * 512 + col] = (__bf16)acc1[r];
    }
  }
  __syncthreads();

  // ---- scores + masked softmax; wave = head, lane = neighbor slot
  {
    const int h = wave, l = lane;
    float a = -3.0e38f;
    if (l < L) {
      float qk = 0.f;
      const float* qb = q_sel + (size_t)n * 1024 + qoff + h * 32;
#pragma unroll
      for (int d = 0; d < 32; ++d) qk += qb[d] * (float)lds_kv[l * 512 + h * 32 + d];
      bool masked = mask[(size_t)n * L + l] != 0;
      a = masked ? -1e10f : qk * scale;
    }
    float mx = a;
#pragma unroll
    for (int o = 16; o > 0; o >>= 1) mx = fmaxf(mx, __shfl_xor(mx, o, 32));
    float e = (l < L) ? __expf(a - mx) : 0.f;
    float s = e;
#pragma unroll
    for (int o = 16; o > 0; o >>= 1) s += __shfl_xor(s, o, 32);
    lds_p[h * 32 + l] = e / s;
  }
  __syncthreads();

  // ---- PV: wave = head, lane = channel
  {
    const int h = wave, d = lane;
    float o = 0.f;
    for (int l = 0; l < L; ++l)
      o += lds_p[h * 32 + l] * (float)lds_kv[l * 512 + 256 + h * 32 + d];
    o_out[(size_t)n * 256 + h * 32 + d] = o;
  }
}

// ---------------------------------------------------------------------------
// LayerNorm over D=256 + convert to bf16. 8 rows per 256-thread block.
// ---------------------------------------------------------------------------
__global__ __launch_bounds__(256)
void ln256_to_bf16_kernel(const float* __restrict__ X, const float* __restrict__ g,
                          const float* __restrict__ b, __bf16* __restrict__ out, int R) {
  int wave = threadIdx.x >> 5, lane = threadIdx.x & 31;
  int row = blockIdx.x * 8 + wave;
  if (row >= R) return;
  const float* xr = X + (size_t)row * 256;
  float v[8];
  float s = 0.f, ss = 0.f;
#pragma unroll
  for (int j = 0; j < 8; ++j) { v[j] = xr[lane * 8 + j]; s += v[j]; ss += v[j] * v[j]; }
#pragma unroll
  for (int o = 16; o > 0; o >>= 1) { s += __shfl_xor(s, o, 32); ss += __shfl_xor(ss, o, 32); }
  float m = s * (1.f / 256.f);
  float rs = rsqrtf(ss * (1.f / 256.f) - m * m + 1e-5f);
#pragma unroll
  for (int j = 0; j < 8; ++j) {
    int c = lane * 8 + j;
    out[(size_t)row * 256 + c] = (__bf16)((v[j] - m) * rs * g[c] + b[c]);
  }
}

__global__ void f32_to_bf16_kernel(const float* __restrict__ in, __bf16* __restrict__ out, int n) {
  int i = blockIdx.x * 256 + threadIdx.x;
  if (i < n) out[i] = (__bf16)in[i];
}

__global__ void silu_mul_bf16_kernel(const float* __restrict__ y1, const float* __restrict__ y3,
                                     __bf16* __restrict__ out, int n) {
  int i = blockIdx.x * 256 + threadIdx.x;
  if (i < n) {
    float a = y1[i];
    float s = a / (1.f + __expf(-a));
    out[i] = (__bf16)(s * y3[i]);
  }
}

__global__ void add_kernel(const float* __restrict__ a, const float* __restrict__ b,
                           float* __restrict__ out, int n) {
  int i = blockIdx.x * 256 + threadIdx.x;
  if (i < n) out[i] = a[i] + b[i];
}

// ---------------------------------------------------------------------------
extern "C" void kernel_launch(void* const* d_in, const int* in_sizes, int n_in,
                              void* d_out, int out_size, void* d_ws, size_t ws_size,
                              hipStream_t stream) {
  (void)in_sizes; (void)n_in; (void)out_size; (void)ws_size;

  const float* x         = (const float*)d_in[0];
  const int*   atype     = (const int*)  d_in[1];
  const float* neigh_l2a = (const float*)d_in[2];
  const int*   mask_l2a  = (const int*)  d_in[3];
  const float* neigh_g2a = (const float*)d_in[4];
  const int*   mask_g2a  = (const int*)  d_in[5];
  const float* neigh_oth = (const float*)d_in[6];
  const int*   mask_oth  = (const int*)  d_in[7];
  const float* neigh_pln = (const float*)d_in[8];
  const int*   mask_pln  = (const int*)  d_in[9];
  const float* wq_g = (const float*)d_in[10];
  const float* wq_b = (const float*)d_in[11];
  const float* wq_w = (const float*)d_in[12];
  const float* kv_l2a_g = (const float*)d_in[13];
  const float* kv_l2a_b = (const float*)d_in[14];
  const float* kv_l2a_w = (const float*)d_in[15];
  const float* kv_g2a_g = (const float*)d_in[16];
  const float* kv_g2a_b = (const float*)d_in[17];
  const float* kv_g2a_w = (const float*)d_in[18];
  const float* kv_oth_g = (const float*)d_in[19];
  const float* kv_oth_b = (const float*)d_in[20];
  const float* kv_oth_w = (const float*)d_in[21];
  const float* kv_pln_g = (const float*)d_in[22];
  const float* kv_pln_b = (const float*)d_in[23];
  const float* kv_pln_w = (const float*)d_in[24];
  const float* afc_w[4] = { (const float*)d_in[25], (const float*)d_in[27],
                            (const float*)d_in[29], (const float*)d_in[31] };
  const float* afc_b[4] = { (const float*)d_in[26], (const float*)d_in[28],
                            (const float*)d_in[30], (const float*)d_in[32] };
  const float* self_w = (const float*)d_in[33];
  const float* self_b = (const float*)d_in[34];
  const float* out_w  = (const float*)d_in[35];
  const float* out_b  = (const float*)d_in[36];
  const float* ffn_g    = (const float*)d_in[37];
  const float* ffn_beta = (const float*)d_in[38];
  const float* ffn_w1 = (const float*)d_in[39];
  const float* ffn_b1 = (const float*)d_in[40];
  const float* ffn_w2 = (const float*)d_in[41];
  const float* ffn_b2 = (const float*)d_in[42];
  const float* ffn_w3 = (const float*)d_in[43];
  const float* ffn_b3 = (const float*)d_in[44];

  // ---- workspace bump allocator ----
  size_t off = 0;
  auto alloc = [&](size_t bytes) -> char* {
    size_t o = (off + 255) & ~(size_t)255;
    off = o + bytes;
    return (char*)d_ws + o;
  };
  __bf16* pw_wq     = (__bf16*)alloc((size_t)3 * 262144 * 2);
  __bf16* pw_kv_l2a = (__bf16*)alloc((size_t)131072 * 2);
  __bf16* pw_kv_g2a = (__bf16*)alloc((size_t)131072 * 2);
  __bf16* pw_kv_oth = (__bf16*)alloc((size_t)3 * 131072 * 2);
  __bf16* pw_kv_pln = (__bf16*)alloc((size_t)3 * 131072 * 2);
  __bf16* pw_afc[4];
  for (int r = 0; r < 4; ++r) pw_afc[r] = (__bf16*)alloc((size_t)3 * 65536 * 2);
  __bf16* pw_self = (__bf16*)alloc((size_t)3 * 65536 * 2);
  __bf16* pw_out  = (__bf16*)alloc((size_t)3 * 327680 * 2);
  __bf16* pw_f1   = (__bf16*)alloc((size_t)3 * 262144 * 2);
  __bf16* pw_f3   = (__bf16*)alloc((size_t)3 * 262144 * 2);
  __bf16* pw_f2   = (__bf16*)alloc((size_t)3 * 262144 * 2);

  __bf16* x_bf16   = (__bf16*)alloc((size_t)NA * 256 * 2);
  __bf16* lnx_bf16 = (__bf16*)alloc((size_t)NA * 256 * 2);
  float*  q_sel    = (float*) alloc((size_t)NA * 1024 * 4);
  float*  o_l2a    = (float*) alloc((size_t)NA * 256 * 4);
  float*  o_g2a    = (float*) alloc((size_t)NA * 256 * 4);
  float*  o_oth    = (float*) alloc((size_t)NA * 256 * 4);
  float*  o_pln    = (float*) alloc((size_t)NA * 256 * 4);
  __bf16* o_bf16   = (__bf16*)alloc((size_t)NA * 256 * 2);
  float*  cat_f32  = (float*) alloc((size_t)NA * 1280 * 4);
  __bf16* cat_bf16 = (__bf16*)alloc((size_t)NA * 1280 * 2);
  float*  h_f32    = (float*) alloc((size_t)NA * 256 * 4);
  __bf16* lnh_bf16 = (__bf16*)alloc((size_t)NA * 256 * 2);
  float*  y1       = (float*) alloc((size_t)NA * 1024 * 4);
  float*  y3       = (float*) alloc((size_t)NA * 1024 * 4);
  __bf16* gbuf     = (__bf16*)alloc((size_t)NA * 1024 * 2);
  float*  ffn_out  = (float*) alloc((size_t)NA * 256 * 4);

  // ---- 1) pack all weights into WMMA fragment order (bf16) ----
  struct PJ { const float* W; __bf16* dst; int K; int Nc; };
  PJ jobs[40];
  int nj = 0;
  for (int t = 0; t < 3; ++t) jobs[nj++] = { wq_w + (size_t)t * 262144, pw_wq + (size_t)t * 262144, 256, 1024 };
  jobs[nj++] = { kv_l2a_w, pw_kv_l2a, 256, 512 };
  jobs[nj++] = { kv_g2a_w, pw_kv_g2a, 256, 512 };
  for (int t = 0; t < 3; ++t) jobs[nj++] = { kv_oth_w + (size_t)t * 131072, pw_kv_oth + (size_t)t * 131072, 256, 512 };
  for (int t = 0; t < 3; ++t) jobs[nj++] = { kv_pln_w + (size_t)t * 131072, pw_kv_pln + (size_t)t * 131072, 256, 512 };
  for (int r = 0; r < 4; ++r)
    for (int t = 0; t < 3; ++t)
      jobs[nj++] = { afc_w[r] + (size_t)t * 65536, pw_afc[r] + (size_t)t * 65536, 256, 256 };
  for (int t = 0; t < 3; ++t) jobs[nj++] = { self_w + (size_t)t * 65536, pw_self + (size_t)t * 65536, 256, 256 };
  for (int t = 0; t < 3; ++t) jobs[nj++] = { out_w + (size_t)t * 327680, pw_out + (size_t)t * 327680, 1280, 256 };
  for (int t = 0; t < 3; ++t) jobs[nj++] = { ffn_w1 + (size_t)t * 262144, pw_f1 + (size_t)t * 262144, 256, 1024 };
  for (int t = 0; t < 3; ++t) jobs[nj++] = { ffn_w3 + (size_t)t * 262144, pw_f3 + (size_t)t * 262144, 256, 1024 };
  for (int t = 0; t < 3; ++t) jobs[nj++] = { ffn_w2 + (size_t)t * 262144, pw_f2 + (size_t)t * 262144, 1024, 256 };
  for (int j = 0; j < nj; ++j) {
    int KT = jobs[j].K / 32, NT = jobs[j].Nc / 16;
    pack_b_kernel<<<KT * NT, 32, 0, stream>>>(jobs[j].W, jobs[j].dst, KT, jobs[j].Nc);
  }

  auto gemm = [&](const __bf16* A, int lda, const __bf16* Bp, int K, int Nc,
                  const float* bias, float* C, int ldc, int M, int act,
                  const int* typ, int tmatch) {
    dim3 grid(M / 64, Nc / 64);
    gemm_bf16_kernel<<<grid, 128, 0, stream>>>(A, lda, Bp, K / 32, bias, C, ldc,
                                               M, act, typ, tmatch);
  };

  // ---- 2) x -> bf16 ----
  f32_to_bf16_kernel<<<(NA * 256 + 255) / 256, 256, 0, stream>>>(x, x_bf16, NA * 256);

  // ---- 3) typed queries: q_sel[n, 0:1024] = LN_t(x) @ wq_w[t]  (t = type[n]) ----
  for (int t = 0; t < 3; ++t) {
    ln256_to_bf16_kernel<<<NA / 8, 256, 0, stream>>>(x, wq_g + t * 256, wq_b + t * 256,
                                                     lnx_bf16, NA);
    gemm(lnx_bf16, 256, pw_wq + (size_t)t * 262144, 256, 1024, nullptr,
         q_sel, 1024, NA, 0, atype, t);
  }

  // ---- 4) fused KV projection + attention per relation ----
  const float inv_sdk = 1.f / sqrtf(32.f);
  float sc_l2a = inv_sdk * (logf(25.f) / logf(32.f));
  float sc_g2a = inv_sdk * (logf(17.f) / logf(32.f));
  float sc_oth = inv_sdk * (logf(33.f) / logf(32.f));
  float sc_pln = inv_sdk * (logf(7.f)  / logf(32.f));
  attn_kv_kernel<<<NA, 256, 0, stream>>>(neigh_l2a, mask_l2a, kv_l2a_g, kv_l2a_b, 0,
                                         pw_kv_l2a, 0, q_sel, 0,   atype, o_l2a, 24, sc_l2a);
  attn_kv_kernel<<<NA, 256, 0, stream>>>(neigh_g2a, mask_g2a, kv_g2a_g, kv_g2a_b, 0,
                                         pw_kv_g2a, 0, q_sel, 256, atype, o_g2a, 16, sc_g2a);
  attn_kv_kernel<<<NA, 256, 0, stream>>>(neigh_oth, mask_oth, kv_oth_g, kv_oth_b, 1,
                                         pw_kv_oth, 131072, q_sel, 512, atype, o_oth, 32, sc_oth);
  attn_kv_kernel<<<NA, 256, 0, stream>>>(neigh_pln, mask_pln, kv_pln_g, kv_pln_b, 1,
                                         pw_kv_pln, 131072, q_sel, 768, atype, o_pln, 6, sc_pln);

  // ---- 5) fc_relu branches -> cat [h_self | h_l2a | h_g2a | h_oth | h_pln] ----
  for (int t = 0; t < 3; ++t)
    gemm(x_bf16, 256, pw_self + (size_t)t * 65536, 256, 256, self_b + t * 256,
         cat_f32 + 0, 1280, NA, 1, atype, t);
  float* obufs[4] = { o_l2a, o_g2a, o_oth, o_pln };
  int coloff[4] = { 256, 512, 768, 1024 };
  for (int r = 0; r < 4; ++r) {
    f32_to_bf16_kernel<<<(NA * 256 + 255) / 256, 256, 0, stream>>>(obufs[r], o_bf16, NA * 256);
    for (int t = 0; t < 3; ++t)
      gemm(o_bf16, 256, pw_afc[r] + (size_t)t * 65536, 256, 256, afc_b[r] + t * 256,
           cat_f32 + coloff[r], 1280, NA, 1, atype, t);
  }

  // ---- 6) out projection: h = cat @ out_w[t] + out_b[t] ----
  f32_to_bf16_kernel<<<(NA * 1280 + 255) / 256, 256, 0, stream>>>(cat_f32, cat_bf16, NA * 1280);
  for (int t = 0; t < 3; ++t)
    gemm(cat_bf16, 1280, pw_out + (size_t)t * 327680, 1280, 256, out_b + t * 256,
         h_f32, 256, NA, 0, atype, t);

  // ---- 7) typed SwiGLU FFN ----
  for (int t = 0; t < 3; ++t) {
    ln256_to_bf16_kernel<<<NA / 8, 256, 0, stream>>>(h_f32, ffn_g + t * 256,
                                                     ffn_beta + t * 256, lnh_bf16, NA);
    gemm(lnh_bf16, 256, pw_f1 + (size_t)t * 262144, 256, 1024, ffn_b1 + t * 1024,
         y1, 1024, NA, 0, nullptr, 0);
    gemm(lnh_bf16, 256, pw_f3 + (size_t)t * 262144, 256, 1024, ffn_b3 + t * 1024,
         y3, 1024, NA, 0, nullptr, 0);
    silu_mul_bf16_kernel<<<(NA * 1024 + 255) / 256, 256, 0, stream>>>(y1, y3, gbuf, NA * 1024);
    gemm(gbuf, 1024, pw_f2 + (size_t)t * 262144, 1024, 256, ffn_b2 + t * 256,
         ffn_out, 256, NA, 0, atype, t);
  }

  // ---- 8) residual ----
  add_kernel<<<(NA * 256 + 255) / 256, 256, 0, stream>>>(h_f32, ffn_out,
                                                         (float*)d_out, NA * 256);
}
